// VGAE_25838523252826
// MI455X (gfx1250) — matrix operand order repaired and checked by reading, not compiled, not run on hardware
//
#include <hip/hip_runtime.h>
#include <hip/hip_bf16.h>
#include <stdint.h>

// ---------------- problem constants (match reference) ----------------
#define NN        8192
#define EE        131072
#define IN_DIM    256
#define INTERM    32
#define LATENT    64
#define MAX_ITERS 32
#define TILES1D   (NN / 16)          // 512
#define LOG63     4.1431347263915326f

typedef __attribute__((ext_vector_type(16))) _Float16 v16h;
typedef __attribute__((ext_vector_type(8)))  _Float16 v8h;
typedef __attribute__((ext_vector_type(8)))  float    v8f;

// ---------------- small utility kernels ----------------
__global__ void fill_f32(float* p, int n, float v) {
    int i = blockIdx.x * blockDim.x + threadIdx.x;
    if (i < n) p[i] = v;
}

// degree: deg starts at 1.0 (self loop); add 1 per in-edge
__global__ void degree_kernel(const int* dst, float* deg) {
    int e = blockIdx.x * blockDim.x + threadIdx.x;
    if (e < EE) atomicAdd(&deg[dst[e]], 1.0f);
}

__global__ void dinv_kernel(const float* deg, float* dinv) {
    int i = blockIdx.x * blockDim.x + threadIdx.x;
    if (i < NN) { float d = deg[i]; dinv[i] = d > 0.f ? rsqrtf(d) : 0.f; }
}

// ---------------- GEMM1: HW1 = X @ W1  [8192,256]x[256,32] ----------------
// W1 staged in LDS (32KB); block = 256 threads = 8 rows x 32 cols.
__global__ void gemm1_kernel(const float* __restrict__ X,
                             const float* __restrict__ W1,
                             float* __restrict__ HW1) {
    __shared__ float w[IN_DIM * INTERM];
    for (int i = threadIdx.x; i < IN_DIM * INTERM; i += 256) w[i] = W1[i];
    __syncthreads();
    int c  = threadIdx.x & 31;
    int rl = threadIdx.x >> 5;
    int r  = blockIdx.x * 8 + rl;
    const float* xr = X + (size_t)r * IN_DIM;
    float acc = 0.f;
#pragma unroll 4
    for (int k = 0; k < IN_DIM; ++k) acc = fmaf(xr[k], w[k * INTERM + c], acc);
    HW1[(size_t)r * INTERM + c] = acc;
}

// edge scatter for layer1: accH[dst] += HW1[src] * dinv[src]*dinv[dst]
__global__ void scatter1_kernel(const int* __restrict__ src, const int* __restrict__ dst,
                                const float* __restrict__ dinv,
                                const float* __restrict__ HW1, float* __restrict__ accH) {
    long long t = (long long)blockIdx.x * blockDim.x + threadIdx.x;
    if (t >= (long long)EE * INTERM) return;
    int e = (int)(t >> 5), c = (int)(t & 31);
    int s = src[e], d = dst[e];
    float nrm = dinv[s] * dinv[d];
    atomicAdd(&accH[(size_t)d * INTERM + c], HW1[(size_t)s * INTERM + c] * nrm);
}

// H = relu(accH + selfloop + b1)
__global__ void finalizeH_kernel(const float* __restrict__ accH, const float* __restrict__ HW1,
                                 const float* __restrict__ dinv, const float* __restrict__ b1,
                                 float* __restrict__ H) {
    int t = blockIdx.x * blockDim.x + threadIdx.x;
    if (t >= NN * INTERM) return;
    int i = t >> 5, c = t & 31;
    float di = dinv[i];
    float v = accH[t] + HW1[t] * di * di + b1[c];
    H[t] = v > 0.f ? v : 0.f;
}

// HWmu = H @ W_mu   [8192,32]x[32,64]
__global__ void gemm_mu_kernel(const float* __restrict__ H, const float* __restrict__ Wmu,
                               float* __restrict__ HWmu) {
    int t = blockIdx.x * blockDim.x + threadIdx.x;
    if (t >= NN * LATENT) return;
    int r = t >> 6, c = t & 63;
    const float* hr = H + (size_t)r * INTERM;
    float acc = 0.f;
#pragma unroll
    for (int k = 0; k < INTERM; ++k) acc = fmaf(hr[k], Wmu[k * LATENT + c], acc);
    HWmu[t] = acc;
}

// HWk = H @ W_k   [8192,32]x[32,1]
__global__ void gemm_k_kernel(const float* __restrict__ H, const float* __restrict__ Wk,
                              float* __restrict__ HWk) {
    int r = blockIdx.x * blockDim.x + threadIdx.x;
    if (r >= NN) return;
    const float* hr = H + (size_t)r * INTERM;
    float acc = 0.f;
#pragma unroll
    for (int k = 0; k < INTERM; ++k) acc = fmaf(hr[k], Wk[k], acc);
    HWk[r] = acc;
}

__global__ void scatter_mu_kernel(const int* __restrict__ src, const int* __restrict__ dst,
                                  const float* __restrict__ dinv,
                                  const float* __restrict__ HWmu, float* __restrict__ accMu) {
    long long t = (long long)blockIdx.x * blockDim.x + threadIdx.x;
    if (t >= (long long)EE * LATENT) return;
    int e = (int)(t >> 6), c = (int)(t & 63);
    int s = src[e], d = dst[e];
    float nrm = dinv[s] * dinv[d];
    atomicAdd(&accMu[(size_t)d * LATENT + c], HWmu[(size_t)s * LATENT + c] * nrm);
}

__global__ void scatter_k_kernel(const int* __restrict__ src, const int* __restrict__ dst,
                                 const float* __restrict__ dinv,
                                 const float* __restrict__ HWk, float* __restrict__ accK) {
    int e = blockIdx.x * blockDim.x + threadIdx.x;
    if (e >= EE) return;
    int s = src[e], d = dst[e];
    atomicAdd(&accK[d], HWk[s] * dinv[s] * dinv[d]);
}

__device__ __forceinline__ float softplus_f(float x) {
    return x > 20.f ? x : log1pf(__expf(x));
}

// finalize mus (normalize, write to out) and kap (softplus, write to out)
__global__ void finalize_mukap_kernel(const float* __restrict__ accMu, const float* __restrict__ HWmu,
                                      const float* __restrict__ bmu,
                                      const float* __restrict__ accK, const float* __restrict__ HWk,
                                      const float* __restrict__ bk,
                                      const float* __restrict__ dinv,
                                      float* __restrict__ mus_out, float* __restrict__ kap_out) {
    int i = blockIdx.x * blockDim.x + threadIdx.x;
    if (i >= NN) return;
    float di2 = dinv[i] * dinv[i];
    float v[LATENT];
    float nrm = 0.f;
#pragma unroll
    for (int c = 0; c < LATENT; ++c) {
        float x = accMu[(size_t)i * LATENT + c] + HWmu[(size_t)i * LATENT + c] * di2 + bmu[c];
        v[c] = x;
        nrm = fmaf(x, x, nrm);
    }
    float inv = 1.f / (sqrtf(nrm) + 1e-12f);
#pragma unroll
    for (int c = 0; c < LATENT; ++c) mus_out[(size_t)i * LATENT + c] = v[c] * inv;
    kap_out[i] = softplus_f(accK[i] + HWk[i] * di2 + bk[0]);
}

// ---------------- stateless hash RNG ----------------
__device__ __forceinline__ uint32_t hash_u32(uint32_t x) {
    x ^= x >> 17; x *= 0xed5ad4bbu;
    x ^= x >> 11; x *= 0xac4c1b51u;
    x ^= x >> 15; x *= 0x31848babu;
    x ^= x >> 14;
    return x;
}
__device__ __forceinline__ float u01(uint32_t node, uint32_t ctr) {
    uint32_t h = hash_u32(node * 0x9e3779b9u + ctr * 0x85ebca6bu + 0x6d2b79f5u);
    return (float)(h >> 8) * (1.0f / 16777216.0f) + (1.0f / 33554432.0f);
}
__device__ __forceinline__ float normal01(uint32_t node, uint32_t* ctr) {
    float u1 = u01(node, (*ctr)++);
    float u2 = u01(node, (*ctr)++);
    return sqrtf(-2.f * __logf(u1)) * __cosf(6.2831853f * u2);
}
// Marsaglia–Tsang gamma(alpha>=1), bounded retries
__device__ __forceinline__ float gamma_mt(float alpha, uint32_t node, uint32_t* ctr) {
    float d = alpha - (1.f / 3.f), c = rsqrtf(9.f * d);
    for (int t = 0; t < 6; ++t) {
        float x = normal01(node, ctr);
        float v = 1.f + c * x; v = v * v * v;
        float u = u01(node, (*ctr)++);
        if (v > 0.f && __logf(u) < 0.5f * x * x + d - d * v + d * __logf(v)) return d * v;
    }
    return d;
}

// vMF sampling per node (Wood's rejection, fixed 32 iters) -> Zh (f16, [N,64])
__global__ void sample_vmf_kernel(const float* __restrict__ mus, const float* __restrict__ kap,
                                  _Float16* __restrict__ Zh) {
    int i = blockIdx.x * blockDim.x + threadIdx.x;
    if (i >= NN) return;
    uint32_t ctr = 1u;
    uint32_t node = (uint32_t)i + 42u * 0x01000193u;

    float k  = kap[i];
    float s  = sqrtf(4.f * k * k + 63.f * 63.f);
    float bb = (-2.f * k + s) / 63.f;
    float aa = (63.f + 2.f * k + s) * 0.25f;
    float dd = 4.f * aa * bb / (1.f + bb) - 63.f * LOG63;

    float w = 0.f;
    bool sampled = false;
#pragma unroll 1
    for (int it = 0; it < MAX_ITERS; ++it) {
        float g1 = gamma_mt(31.5f, node, &ctr);
        float g2 = gamma_mt(31.5f, node, &ctr);
        float eps = g1 / (g1 + g2);
        float denom = 1.f - (1.f - bb) * eps;
        float w_prop = (1.f - (1.f + bb) * eps) / denom;
        float t = 2.f * aa * bb / denom;
        float u = u01(node, ctr++);
        bool acc = (63.f * __logf(t) - t + dd) >= __logf(u);
        if (!sampled) { w = w_prop; sampled = acc; }
    }

    // tangential direction v in R^63, normalized
    float v[LATENT - 1];
    float nv = 0.f;
#pragma unroll 1
    for (int j = 0; j < LATENT - 1; ++j) { float x = normal01(node, &ctr); v[j] = x; nv = fmaf(x, x, nv); }
    float inv = rsqrtf(nv);
    float sw = sqrtf(fmaxf(1.f - w * w, 0.f));

    float z[LATENT];
    z[0] = w;
#pragma unroll
    for (int j = 0; j < LATENT - 1; ++j) z[j + 1] = sw * v[j] * inv;

    // householder e1 -> mu
    float u0[LATENT];
    float nu = 0.f;
#pragma unroll
    for (int j = 0; j < LATENT; ++j) {
        float m = mus[(size_t)i * LATENT + j];
        float uj = (j == 0 ? 1.f : 0.f) - m;
        u0[j] = uj;
        nu = fmaf(uj, uj, nu);
    }
    float invu = 1.f / (sqrtf(nu) + 1e-8f);
    float dot = 0.f;
#pragma unroll
    for (int j = 0; j < LATENT; ++j) { u0[j] *= invu; dot = fmaf(z[j], u0[j], dot); }
#pragma unroll
    for (int j = 0; j < LATENT; ++j) {
        float zz = z[j] - 2.f * dot * u0[j];
        Zh[(size_t)i * LATENT + j] = (_Float16)zz;
    }
}

// ---------------- ZZt = Z @ Z^T via WMMA f16 (f32 accum) ----------------
// one wave per 16x16 output tile, K = 64 = 2 x wmma_f32_16x16x32_f16.
// A fragment (16-bit A 16x32, contiguous k order of this kernel, same as B): lane l (<16 -> rows, half h=l/16):
//   v0..v3 hold K = 16h+0..7 ; v4..v7 hold K = 16h+8..15  -> two 16B loads.
// B fragment (32x16): lane = column, v0..v7 hold K = 16h+0..15 -> one 32B load.
__global__ void zzt_wmma_kernel(const _Float16* __restrict__ Zh, float* __restrict__ out) {
    int lane = threadIdx.x & 31;
    int wave = threadIdx.x >> 5;
    int tile = blockIdx.x * 8 + wave;
    int ti = tile >> 9;           // /512
    int tj = tile & (TILES1D - 1);
    int l16 = lane & 15;
    int half = lane >> 4;

    const _Float16* Arow = Zh + ((size_t)(ti * 16 + l16)) * LATENT;
    const _Float16* Brow = Zh + ((size_t)(tj * 16 + l16)) * LATENT;

    v8f c;
#pragma unroll
    for (int r = 0; r < 8; ++r) c[r] = 0.f;

#pragma unroll
    for (int kc = 0; kc < LATENT; kc += 32) {
        v8h alo = *(const v8h*)(Arow + kc + half * 16);
        v8h ahi = *(const v8h*)(Arow + kc + 8 + half * 16);
        v16h a, b;
        b = *(const v16h*)(Brow + kc + half * 16);
#pragma unroll
        for (int q = 0; q < 8; ++q) { a[q] = alo[q]; a[8 + q] = ahi[q]; }
        c = __builtin_amdgcn_wmma_f32_16x16x32_f16(
                /*neg_a=*/false, a, /*neg_b=*/false, b,
                /*c_mod=*/(short)0, c, /*reuse_a=*/false, /*reuse_b=*/false);
    }

    // C/D layout: VGPR r, lanes 0-15 -> M=r, lanes 16-31 -> M=8+r; N = l16.
    size_t col = (size_t)(tj * 16 + l16);
#pragma unroll
    for (int r = 0; r < 8; ++r) {
        size_t row = (size_t)(ti * 16 + half * 8 + r);
        __builtin_nontemporal_store(c[r], out + row * NN + col);
    }
}

// ---------------- launcher ----------------
extern "C" void kernel_launch(void* const* d_in, const int* in_sizes, int n_in,
                              void* d_out, int out_size, void* d_ws, size_t ws_size,
                              hipStream_t stream) {
    (void)in_sizes; (void)n_in; (void)out_size; (void)ws_size;
    const float* X    = (const float*)d_in[0];
    const int*   eidx = (const int*)d_in[1];   // [2,E] flat: src then dst
    const float* W1   = (const float*)d_in[2];
    const float* b1   = (const float*)d_in[3];
    const float* Wmu  = (const float*)d_in[4];
    const float* bmu  = (const float*)d_in[5];
    const float* Wk   = (const float*)d_in[6];
    const float* bk   = (const float*)d_in[7];

    const int* src = eidx;
    const int* dst = eidx + EE;

    // output layout: ZZt [N*N], mus [N*64], kap [N]
    float* out_zzt = (float*)d_out;
    float* out_mus = out_zzt + (size_t)NN * NN;
    float* out_kap = out_mus + (size_t)NN * LATENT;

    // workspace carve-up (256B aligned blocks)
    char* ws = (char*)d_ws;
    size_t off = 0;
    auto carve = [&](size_t bytes) { void* p = ws + off; off = (off + bytes + 255) & ~(size_t)255; return p; };
    float*    deg   = (float*)carve(NN * 4);
    float*    dinv  = (float*)carve(NN * 4);
    float*    HW1   = (float*)carve((size_t)NN * INTERM * 4);
    float*    accH  = (float*)carve((size_t)NN * INTERM * 4);
    float*    Hbuf  = (float*)carve((size_t)NN * INTERM * 4);
    float*    HWmu  = (float*)carve((size_t)NN * LATENT * 4);
    float*    accMu = (float*)carve((size_t)NN * LATENT * 4);
    float*    HWk   = (float*)carve(NN * 4);
    float*    accK  = (float*)carve(NN * 4);
    _Float16* Zh    = (_Float16*)carve((size_t)NN * LATENT * 2);

    const int T = 256;
    // init accumulators every call (graph-replay safe)
    fill_f32<<<(NN + T - 1) / T, T, 0, stream>>>(deg, NN, 1.0f);                    // self-loop
    fill_f32<<<((NN * INTERM) + T - 1) / T, T, 0, stream>>>(accH, NN * INTERM, 0.f);
    fill_f32<<<((NN * LATENT) + T - 1) / T, T, 0, stream>>>(accMu, NN * LATENT, 0.f);
    fill_f32<<<(NN + T - 1) / T, T, 0, stream>>>(accK, NN, 0.f);

    degree_kernel<<<(EE + T - 1) / T, T, 0, stream>>>(dst, deg);
    dinv_kernel<<<(NN + T - 1) / T, T, 0, stream>>>(deg, dinv);

    gemm1_kernel<<<NN / 8, T, 0, stream>>>(X, W1, HW1);
    scatter1_kernel<<<(int)(((long long)EE * INTERM + T - 1) / T), T, 0, stream>>>(src, dst, dinv, HW1, accH);
    finalizeH_kernel<<<((NN * INTERM) + T - 1) / T, T, 0, stream>>>(accH, HW1, dinv, b1, Hbuf);

    gemm_mu_kernel<<<((NN * LATENT) + T - 1) / T, T, 0, stream>>>(Hbuf, Wmu, HWmu);
    gemm_k_kernel<<<(NN + T - 1) / T, T, 0, stream>>>(Hbuf, Wk, HWk);
    scatter_mu_kernel<<<(int)(((long long)EE * LATENT + T - 1) / T), T, 0, stream>>>(src, dst, dinv, HWmu, accMu);
    scatter_k_kernel<<<(EE + T - 1) / T, T, 0, stream>>>(src, dst, dinv, HWk, accK);
    finalize_mukap_kernel<<<(NN + T - 1) / T, T, 0, stream>>>(accMu, HWmu, bmu, accK, HWk, bk, dinv,
                                                             out_mus, out_kap);

    sample_vmf_kernel<<<(NN + T - 1) / T, T, 0, stream>>>(out_mus, out_kap, Zh);

    // 512x512 tiles, 8 waves (tiles) per 256-thread block
    zzt_wmma_kernel<<<(TILES1D * TILES1D) / 8, T, 0, stream>>>(Zh, out_zzt);
}
